// MultiheadAttention_mod_54898271977831
// MI455X (gfx1250) — matrix operand order repaired and checked by reading, not compiled, not run on hardware
//
#include <hip/hip_runtime.h>
#include <stdint.h>

typedef __bf16 bf16_t;
typedef __bf16 v16bf __attribute__((ext_vector_type(16)));
typedef float  v8f   __attribute__((ext_vector_type(8)));
typedef unsigned int uu4 __attribute__((ext_vector_type(4)));

union BF16Reg { v16bf v; uu4 q[2]; };

#define WMMA_BF16(A, B, C) \
  __builtin_amdgcn_wmma_f32_16x16x32_bf16(false, (A), false, (B), (short)0, (C), false, false)

static constexpr int kS    = 2048;
static constexpr int kB    = 2;
static constexpr int kDIN  = 512;
static constexpr int kH    = 16;
static constexpr int kHD   = 64;
static constexpr int kE    = 1024;          // H*HD
static constexpr int kROWS = kS * kB;       // 4096 GEMM rows
static constexpr int kNC   = 3 * kE;        // 3072 qkv columns

__device__ __forceinline__ v8f zero8() {
  v8f z = {0.f, 0.f, 0.f, 0.f, 0.f, 0.f, 0.f, 0.f};
  return z;
}

// ---------------------------------------------------------------------------
// Prep: convert x -> bf16, build concatenated [conn_w; qkv_w] bf16 weight,
// o_w -> bf16, concatenated bias (f32).
// ---------------------------------------------------------------------------
__global__ void prep_kernel(const float* __restrict__ x,
                            const float* __restrict__ qkv_w,
                            const float* __restrict__ qkv_b,
                            const float* __restrict__ conn_w,
                            const float* __restrict__ conn_b,
                            const float* __restrict__ o_w,
                            bf16_t* __restrict__ xbf,
                            bf16_t* __restrict__ wcat,
                            bf16_t* __restrict__ owbf,
                            float*  __restrict__ biascat) {
  const int i      = blockIdx.x * blockDim.x + threadIdx.x;
  const int stride = gridDim.x * blockDim.x;
  const int NX = kROWS * kDIN;   // 2097152
  const int NW = kNC * kDIN;     // 1572864
  const int NO = kHD * kE;       // 65536
  for (int t = i; t < NX; t += stride) xbf[t] = (bf16_t)x[t];
  for (int t = i; t < NW; t += stride) {
    int n = t >> 9, k = t & 511;
    float w = (n < kHD) ? conn_w[n * kDIN + k] : qkv_w[(n - kHD) * kDIN + k];
    wcat[t] = (bf16_t)w;
  }
  for (int t = i; t < NO; t += stride) owbf[t] = (bf16_t)o_w[t];
  for (int t = i; t < kNC; t += stride)
    biascat[t] = (t < kHD) ? conn_b[t] : qkv_b[t - kHD];
}

// ---------------------------------------------------------------------------
// GEMM1: Y(4096x3072) = Xbf(4096x512) * Wcat^T + bias, output bf16.
// Per wave: 16(M) x 64(N) tile, 4 accumulators, K step 32.
// Grid: (3072/64, 4096/128), block 256 (8 waves stacked in M).
// ---------------------------------------------------------------------------
__global__ __launch_bounds__(256) void gemm_qkv_kernel(
    const bf16_t* __restrict__ xbf, const bf16_t* __restrict__ wcat,
    const float* __restrict__ biascat, bf16_t* __restrict__ y) {
  const int lane = threadIdx.x & 31;
  const int wave = threadIdx.x >> 5;
  const int lm = lane & 15, lh = lane >> 4;
  const int mbase = blockIdx.y * 128 + wave * 16;
  const int nbase = blockIdx.x * 64;

  v8f acc0 = zero8(), acc1 = zero8(), acc2 = zero8(), acc3 = zero8();

  const bf16_t* arow = xbf + (size_t)(mbase + lm) * kDIN + lh * 16;
  const bf16_t* br0  = wcat + (size_t)(nbase +  0 + lm) * kDIN + lh * 16;
  const bf16_t* br1  = wcat + (size_t)(nbase + 16 + lm) * kDIN + lh * 16;
  const bf16_t* br2  = wcat + (size_t)(nbase + 32 + lm) * kDIN + lh * 16;
  const bf16_t* br3  = wcat + (size_t)(nbase + 48 + lm) * kDIN + lh * 16;

#pragma unroll 4
  for (int k0 = 0; k0 < kDIN; k0 += 32) {
    __builtin_prefetch(arow + k0 + 64, 0, 0);
    __builtin_prefetch(br0 + k0 + 64, 0, 0);
    BF16Reg a, b0, b1, b2, b3;
    a.q[0]  = *(const uu4*)(arow + k0);      a.q[1]  = *(const uu4*)(arow + k0 + 8);
    b0.q[0] = *(const uu4*)(br0 + k0);       b0.q[1] = *(const uu4*)(br0 + k0 + 8);
    b1.q[0] = *(const uu4*)(br1 + k0);       b1.q[1] = *(const uu4*)(br1 + k0 + 8);
    b2.q[0] = *(const uu4*)(br2 + k0);       b2.q[1] = *(const uu4*)(br2 + k0 + 8);
    b3.q[0] = *(const uu4*)(br3 + k0);       b3.q[1] = *(const uu4*)(br3 + k0 + 8);
    acc0 = WMMA_BF16(a.v, b0.v, acc0);
    acc1 = WMMA_BF16(a.v, b1.v, acc1);
    acc2 = WMMA_BF16(a.v, b2.v, acc2);
    acc3 = WMMA_BF16(a.v, b3.v, acc3);
  }

  v8f acc[4] = {acc0, acc1, acc2, acc3};
#pragma unroll
  for (int t = 0; t < 4; ++t) {
    const int col = nbase + t * 16 + lm;
    const float bias = biascat[col];
#pragma unroll
    for (int g = 0; g < 8; ++g) {
      const int row = mbase + lh * 8 + g;  // C layout: M = (lane>>4)*8 + g
      y[(size_t)row * kNC + col] = (bf16_t)(acc[t][g] + bias);
    }
  }
}

// ---------------------------------------------------------------------------
// Flash attention per (b2, h, q-block). Wave owns 16 q-rows, WG owns 128.
// K/V 32x64 tiles staged cooperatively into LDS via global_load_async_to_lds.
// Logits computed as S^T = K * Q^T so softmax rows are lane-local; the
// C-layout of the two S^T tiles maps directly onto the A-layout of P for
// the P*V WMMA (no transpose traffic).
// Grid: (16 qblocks, 16 heads, 2 batches), block 256.
// ---------------------------------------------------------------------------
__global__ __launch_bounds__(256) void attn_kernel(const bf16_t* __restrict__ y,
                                                   bf16_t* __restrict__ vals) {
  __shared__ bf16_t shK[32][64];
  __shared__ bf16_t shV[32][64];

  const int lane = threadIdx.x & 31;
  const int wave = threadIdx.x >> 5;
  const int lm = lane & 15, lh = lane >> 4;
  const int b2 = blockIdx.z;
  const int h  = blockIdx.y;
  const int qrow = blockIdx.x * 128 + wave * 16;  // s2 base for this wave
  const int rowbase = b2 * kS;                    // Y row offset for batch b2

  // Preload Q^T as two B-operands (d-chunks 0..31 and 32..63).
  BF16Reg bq0, bq1;
  {
    const bf16_t* qp = y + (size_t)(rowbase + qrow + lm) * kNC + h * 192 + lh * 8;
    bq0.q[0] = *(const uu4*)(qp);      bq0.q[1] = *(const uu4*)(qp + 16);
    bq1.q[0] = *(const uu4*)(qp + 32); bq1.q[1] = *(const uu4*)(qp + 48);
  }

  v8f acc[4] = {zero8(), zero8(), zero8(), zero8()};
  float m_run = -1e30f, l_run = 0.f;

  // Cooperative async-staging addresses: 256 threads x 16B = 4KB per matrix.
  const int tid = threadIdx.x;
  const int ck = tid >> 3;         // k row 0..31
  const int cc = (tid & 7) * 8;    // d chunk 0,8,..,56
  const uint64_t gK =
      (uint64_t)(const void*)(y + (size_t)(rowbase + ck) * kNC + h * 192 + 64 + cc);
  const uint64_t gV = gK + (uint64_t)(64 * sizeof(bf16_t));
  const unsigned ldsK = (unsigned)(uintptr_t)&shK[ck][cc];
  const unsigned ldsV = (unsigned)(uintptr_t)&shV[ck][cc];

  for (int kt = 0; kt < kS; kt += 32) {
    const uint64_t off = (uint64_t)kt * kNC * sizeof(bf16_t);
    const uint64_t aK = gK + off, aV = gV + off;
    asm volatile("global_load_async_to_lds_b128 %0, %1, off" ::"v"(ldsK), "v"(aK)
                 : "memory");
    asm volatile("global_load_async_to_lds_b128 %0, %1, off" ::"v"(ldsV), "v"(aV)
                 : "memory");
    asm volatile("s_wait_asynccnt 0" ::: "memory");
    __syncthreads();

    // ---- logits: S^T tiles (16k x 16q), two k-subtiles, K(=d) step 32 ----
    v8f st0 = zero8(), st1 = zero8();
    {
      const bf16_t* kp = &shK[lm][0];
      BF16Reg a0, a1;
      a0.q[0] = *(const uu4*)(kp + lh * 8);       a0.q[1] = *(const uu4*)(kp + 16 + lh * 8);
      a1.q[0] = *(const uu4*)(kp + 32 + lh * 8);  a1.q[1] = *(const uu4*)(kp + 48 + lh * 8);
      st0 = WMMA_BF16(a0.v, bq0.v, st0);
      st0 = WMMA_BF16(a1.v, bq1.v, st0);
    }
    {
      const bf16_t* kp = &shK[16 + lm][0];
      BF16Reg a0, a1;
      a0.q[0] = *(const uu4*)(kp + lh * 8);       a0.q[1] = *(const uu4*)(kp + 16 + lh * 8);
      a1.q[0] = *(const uu4*)(kp + 32 + lh * 8);  a1.q[1] = *(const uu4*)(kp + 48 + lh * 8);
      st1 = WMMA_BF16(a0.v, bq0.v, st1);
      st1 = WMMA_BF16(a1.v, bq1.v, st1);
    }

    // ---- online softmax (lane holds q = lane&15; pair (l, l+16) splits k) ----
    float p0[8], p1[8];
    float tmax = -1e30f;
#pragma unroll
    for (int g = 0; g < 8; ++g) {
      p0[g] = st0[g] * 0.125f;
      p1[g] = st1[g] * 0.125f;
      tmax = fmaxf(tmax, fmaxf(p0[g], p1[g]));
    }
    tmax = fmaxf(tmax, __shfl_xor(tmax, 16, 32));
    const float m_new = fmaxf(m_run, tmax);
    const float corr = __expf(m_run - m_new);
    float psum = 0.f;
#pragma unroll
    for (int g = 0; g < 8; ++g) {
      p0[g] = __expf(p0[g] - m_new); psum += p0[g];
      p1[g] = __expf(p1[g] - m_new); psum += p1[g];
    }
    psum += __shfl_xor(psum, 16, 32);
    l_run = l_run * corr + psum;
    m_run = m_new;

    // Rescale O accumulators; redistribute corr(q) to C-layout rows.
#pragma unroll
    for (int g = 0; g < 8; ++g) {
      const float cg = __shfl(corr, lh * 8 + g, 32);
      acc[0][g] *= cg; acc[1][g] *= cg; acc[2][g] *= cg; acc[3][g] *= cg;
    }

    // ---- P (C-layout of S^T tiles) == A-layout of P(16q x 32k) ----
    BF16Reg ap;
#pragma unroll
    for (int g = 0; g < 8; ++g) {
      ap.v[g]     = (bf16_t)p0[g];
      ap.v[8 + g] = (bf16_t)p1[g];
    }

    // ---- O += P * V  (4 d-tiles of 16) ----
#pragma unroll
    for (int t = 0; t < 4; ++t) {
      BF16Reg bv;
#pragma unroll
      for (int j = 0; j < 16; ++j) bv.v[j] = shV[(j < 8) ? (lh * 8 + j) : (16 + lh * 8 + j - 8)][t * 16 + lm];
      acc[t] = WMMA_BF16(ap.v, bv.v, acc[t]);
    }
    __syncthreads();
  }

  // ---- finalize: O /= l, store to vals[(b2*S + s2), h*64 + d] as bf16 ----
  const float inv = 1.f / l_run;
#pragma unroll
  for (int g = 0; g < 8; ++g) {
    const float ig = __shfl(inv, lh * 8 + g, 32);
    const int row = rowbase + qrow + lh * 8 + g;  // rb = b2*2048 + s2
#pragma unroll
    for (int t = 0; t < 4; ++t)
      vals[(size_t)row * kE + h * 64 + t * 16 + lm] = (bf16_t)(acc[t][g] * ig);
  }
}

// ---------------------------------------------------------------------------
// GEMM3: out = vals(4096x1024) * o_w^T(1024x64) + o_b, with row permutation
// rb=(b2*2048+s2) -> r_out = 2*s2 + b2. Output f32.
// Grid: 32 blocks x 256 threads (wave: 16M x 64N, full N).
// ---------------------------------------------------------------------------
__global__ __launch_bounds__(256) void gemm_out_kernel(
    const bf16_t* __restrict__ vals, const bf16_t* __restrict__ owbf,
    const float* __restrict__ o_b, float* __restrict__ out) {
  const int lane = threadIdx.x & 31;
  const int wave = threadIdx.x >> 5;
  const int lm = lane & 15, lh = lane >> 4;
  const int mbase = blockIdx.x * 128 + wave * 16;

  v8f acc0 = zero8(), acc1 = zero8(), acc2 = zero8(), acc3 = zero8();

  const bf16_t* arow = vals + (size_t)(mbase + lm) * kE + lh * 16;
  const bf16_t* br0  = owbf + (size_t)( 0 + lm) * kE + lh * 16;
  const bf16_t* br1  = owbf + (size_t)(16 + lm) * kE + lh * 16;
  const bf16_t* br2  = owbf + (size_t)(32 + lm) * kE + lh * 16;
  const bf16_t* br3  = owbf + (size_t)(48 + lm) * kE + lh * 16;

#pragma unroll 4
  for (int k0 = 0; k0 < kE; k0 += 32) {
    __builtin_prefetch(arow + k0 + 64, 0, 0);
    BF16Reg a, b0, b1, b2, b3;
    a.q[0]  = *(const uu4*)(arow + k0);  a.q[1]  = *(const uu4*)(arow + k0 + 8);
    b0.q[0] = *(const uu4*)(br0 + k0);   b0.q[1] = *(const uu4*)(br0 + k0 + 8);
    b1.q[0] = *(const uu4*)(br1 + k0);   b1.q[1] = *(const uu4*)(br1 + k0 + 8);
    b2.q[0] = *(const uu4*)(br2 + k0);   b2.q[1] = *(const uu4*)(br2 + k0 + 8);
    b3.q[0] = *(const uu4*)(br3 + k0);   b3.q[1] = *(const uu4*)(br3 + k0 + 8);
    acc0 = WMMA_BF16(a.v, b0.v, acc0);
    acc1 = WMMA_BF16(a.v, b1.v, acc1);
    acc2 = WMMA_BF16(a.v, b2.v, acc2);
    acc3 = WMMA_BF16(a.v, b3.v, acc3);
  }

  v8f acc[4] = {acc0, acc1, acc2, acc3};
#pragma unroll
  for (int t = 0; t < 4; ++t) {
    const int col = t * 16 + lm;
    const float bias = o_b[col];
#pragma unroll
    for (int g = 0; g < 8; ++g) {
      const int rb = mbase + lh * 8 + g;
      const int rout = ((rb & 2047) << 1) | (rb >> 11);  // (s2,b2) interleave
      out[(size_t)rout * kHD + col] = acc[t][g] + bias;
    }
  }
}

// ---------------------------------------------------------------------------
// Workspace layout (bytes):
//   [0)          Xbf    4096*512*2  = 4,194,304
//   [4194304)    Wcat   3072*512*2  = 3,145,728
//   [7340032)    OWbf   64*1024*2   =   131,072
//   [7471104)    bias   3072*4      =    12,288
//   [7483392)    Y      4096*3072*2 = 25,165,824
//   [32649216)   VALS   4096*1024*2 =  8,388,608   (total ~39.1 MB)
// ---------------------------------------------------------------------------
extern "C" void kernel_launch(void* const* d_in, const int* in_sizes, int n_in,
                              void* d_out, int out_size, void* d_ws, size_t ws_size,
                              hipStream_t stream) {
  const float* x      = (const float*)d_in[0];
  const float* qkv_w  = (const float*)d_in[1];
  const float* qkv_b  = (const float*)d_in[2];
  const float* conn_w = (const float*)d_in[3];
  const float* conn_b = (const float*)d_in[4];
  const float* o_w    = (const float*)d_in[5];
  const float* o_b    = (const float*)d_in[6];

  char* ws = (char*)d_ws;
  bf16_t* xbf     = (bf16_t*)(ws + 0);
  bf16_t* wcat    = (bf16_t*)(ws + 4194304);
  bf16_t* owbf    = (bf16_t*)(ws + 7340032);
  float*  biascat = (float*) (ws + 7471104);
  bf16_t* y       = (bf16_t*)(ws + 7483392);
  bf16_t* vals    = (bf16_t*)(ws + 32649216);

  prep_kernel<<<2048, 256, 0, stream>>>(x, qkv_w, qkv_b, conn_w, conn_b, o_w,
                                        xbf, wcat, owbf, biascat);
  gemm_qkv_kernel<<<dim3(kNC / 64, kROWS / 128), 256, 0, stream>>>(xbf, wcat,
                                                                   biascat, y);
  attn_kernel<<<dim3(kS / 128, kH, kB), 256, 0, stream>>>(y, vals);
  gemm_out_kernel<<<kROWS / 128, 256, 0, stream>>>(vals, owbf, o_b,
                                                   (float*)d_out);
}
